// Model3_86449101734133
// MI455X (gfx1250) — matrix operand order repaired and checked by reading, not compiled, not run on hardware
//
#include <hip/hip_runtime.h>
#include <cstdint>
#include <cstddef>

// ---------------------------------------------------------------------------
// MI455X (gfx1250, wave32) implementation. All large matmuls use
// v_wmma_f32_16x16x32_bf16 (bf16 operands, f32 accumulate).
// GEMM wave tile: 32(M) x 64(N): 8 WMMA per 12 b128 loads per K-step of 32.
// Workspace requirement: ~230 MB.
// ---------------------------------------------------------------------------

typedef __bf16 BF;
typedef __attribute__((ext_vector_type(16))) BF    v16bf;
typedef __attribute__((ext_vector_type(8)))  BF    v8bf;
typedef __attribute__((ext_vector_type(8)))  float v8f;

namespace {
constexpr int BSZ = 4096;   // batch
constexpr int NIN = 35;     // feature width
constexpr int KP  = 64;     // padded feature width
constexpr int H   = 1024;
constexpr int H3  = 3072;
constexpr int CD  = 1024;   // C*D
constexpr int NC  = 16;     // C
enum { ACT_NONE = 0, ACT_ELU = 1, ACT_MISH = 2 };
}

__device__ __forceinline__ BF tobf(float f) {
  unsigned u = __builtin_bit_cast(unsigned, f);
  unsigned r = u + 0x7fffu + ((u >> 16) & 1u);   // round-to-nearest-even
  unsigned short h = (unsigned short)(r >> 16);
  return __builtin_bit_cast(BF, h);
}

__device__ __forceinline__ v16bf load_afrag(const BF* p) {
  v8bf lo = *(const v8bf*)p;          // K = kh*16 .. +7
  v8bf hi = *(const v8bf*)(p + 8);    // K = kh*16 + 8 .. +7
  return __builtin_shufflevector(lo, hi, 0,1,2,3,4,5,6,7,8,9,10,11,12,13,14,15);
}

// ---------------------------------------------------------------------------
// WMMA GEMM: D[M,N] = act(A[M,K] @ Bt[N,K]^T + bias), M = BSZ fixed.
// block = 128 threads (4 waves); wave -> 32x64 tile; grid = (N/64, BSZ/128).
// scale != null: D[row,col] += scale[row*8] * (acc + bias)   (MoE combine)
// ---------------------------------------------------------------------------
__global__ __launch_bounds__(128) void gemm_bf16(
    const BF* __restrict__ A, int lda,
    const BF* __restrict__ Bm, int ldb,
    const float* __restrict__ bias,
    float* __restrict__ Df, BF* __restrict__ Dbf, int ldd,
    const float* __restrict__ scale,
    int K, int nvalid, int act)
{
  const int lane = threadIdx.x & 31;
  const int wave = threadIdx.x >> 5;
  const int row0 = (blockIdx.y * 4 + wave) * 32;
  const int col0 = blockIdx.x * 64;
  const int r  = lane & 15;
  const int kh = lane >> 4;                 // 0 / 1 (lane half)

  const BF* ap0 = A  + (size_t)(row0 + r) * lda + kh * 16;       // A frag rows 0..15
  const BF* ap1 = A  + (size_t)(row0 + 16 + r) * lda + kh * 16;  // A frag rows 16..31
  const BF* bp  = Bm + (size_t)(col0 + r) * ldb + kh * 16;       // B frag: 16 contig K
  const size_t bstep = (size_t)ldb * 16;

  v8f acc[2][4] = {};
  for (int k0 = 0; k0 < K; k0 += 32) {
    v16bf fa0 = load_afrag(ap0 + k0);
    v16bf fa1 = load_afrag(ap1 + k0);
    v16bf f0 = *(const v16bf*)(bp + k0);
    v16bf f1 = *(const v16bf*)(bp + bstep + k0);
    v16bf f2 = *(const v16bf*)(bp + 2 * bstep + k0);
    v16bf f3 = *(const v16bf*)(bp + 3 * bstep + k0);
    acc[0][0] = __builtin_amdgcn_wmma_f32_16x16x32_bf16(false, fa0, false, f0, (short)0, acc[0][0], false, false);
    acc[0][1] = __builtin_amdgcn_wmma_f32_16x16x32_bf16(false, fa0, false, f1, (short)0, acc[0][1], false, false);
    acc[0][2] = __builtin_amdgcn_wmma_f32_16x16x32_bf16(false, fa0, false, f2, (short)0, acc[0][2], false, false);
    acc[0][3] = __builtin_amdgcn_wmma_f32_16x16x32_bf16(false, fa0, false, f3, (short)0, acc[0][3], false, false);
    acc[1][0] = __builtin_amdgcn_wmma_f32_16x16x32_bf16(false, fa1, false, f0, (short)0, acc[1][0], false, false);
    acc[1][1] = __builtin_amdgcn_wmma_f32_16x16x32_bf16(false, fa1, false, f1, (short)0, acc[1][1], false, false);
    acc[1][2] = __builtin_amdgcn_wmma_f32_16x16x32_bf16(false, fa1, false, f2, (short)0, acc[1][2], false, false);
    acc[1][3] = __builtin_amdgcn_wmma_f32_16x16x32_bf16(false, fa1, false, f3, (short)0, acc[1][3], false, false);
  }

  #pragma unroll
  for (int mh = 0; mh < 2; mh++) {
    #pragma unroll
    for (int tt = 0; tt < 4; tt++) {
      int col = col0 + tt * 16 + r;
      if (col >= nvalid) continue;
      float bv = bias ? bias[col] : 0.f;
      #pragma unroll
      for (int i = 0; i < 8; i++) {
        int row = row0 + mh * 16 + kh * 8 + i;   // C/D layout: VGPR i -> M = 8*half + i
        float v = acc[mh][tt][i] + bv;
        if (act == ACT_ELU)       v = v > 0.f ? v : expm1f(v);
        else if (act == ACT_MISH) { float sp = (v > 20.f) ? v : log1pf(expf(v)); v = v * tanhf(sp); }
        size_t o = (size_t)row * ldd + col;
        if (scale) { Df[o] += scale[(size_t)row * 8] * v; }
        else {
          if (Df)  Df[o]  = v;
          if (Dbf) Dbf[o] = tobf(v);
        }
      }
    }
  }
}

// ---------------------------------------------------------------------------
// LayerNorm over 1024 cols; optional preceding ELU; optional affine; f32/bf16 out.
// ---------------------------------------------------------------------------
__global__ __launch_bounds__(256) void ln1024(
    const float* __restrict__ X, int ldx,
    const float* __restrict__ g, const float* __restrict__ bta,
    float* __restrict__ Yf, BF* __restrict__ Ybf, int ldo, int pre_elu)
{
  int row = blockIdx.x, tid = threadIdx.x;
  const float* xp = X + (size_t)row * ldx;
  float v[4], s = 0.f, sq = 0.f;
  #pragma unroll
  for (int i = 0; i < 4; i++) {
    float t = xp[tid + i * 256];
    if (pre_elu) t = t > 0.f ? t : expm1f(t);
    v[i] = t; s += t; sq += t * t;
  }
  __shared__ float sh[512];
  sh[tid] = s; sh[tid + 256] = sq;
  __syncthreads();
  for (int o = 128; o > 0; o >>= 1) {
    if (tid < o) { sh[tid] += sh[tid + o]; sh[tid + 256] += sh[tid + 256 + o]; }
    __syncthreads();
  }
  float mean = sh[0]   * (1.f / 1024.f);
  float var  = sh[256] * (1.f / 1024.f) - mean * mean;
  float inv  = rsqrtf(var + 1e-5f);
  #pragma unroll
  for (int i = 0; i < 4; i++) {
    int c = tid + i * 256;
    float y = (v[i] - mean) * inv;
    if (g) y = y * g[c] + bta[c];
    size_t o = (size_t)row * ldo + c;
    if (Yf)  Yf[o]  = y;
    if (Ybf) Ybf[o] = tobf(y);
  }
}

// x_vec = x[:, :35] - x[:, 315:350]; t_vec = t[:, 315:350] - x[:, 315:350]; pad to 64
__global__ void prep_vec(const float* __restrict__ x, const float* __restrict__ t,
                         float* XV, float* TV, BF* XVB, BF* TVB)
{
  int idx = blockIdx.x * blockDim.x + threadIdx.x;   // BSZ*64
  int b = idx >> 6, k = idx & 63;
  float xv = 0.f, tv = 0.f;
  if (k < NIN) {
    float xr = x[(size_t)b * 350 + 315 + k];
    xv = x[(size_t)b * 350 + k] - xr;
    tv = t[(size_t)b * 350 + 315 + k] - xr;
  }
  XV[idx] = xv; TV[idx] = tv; XVB[idx] = tobf(xv); TVB[idx] = tobf(tv);
}

__global__ void conv_bf(const float* __restrict__ in, BF* __restrict__ out, int n) {
  int i = blockIdx.x * 256 + threadIdx.x;
  if (i < n) out[i] = tobf(in[i]);
}

// e_fc1_w [1024,35] -> bf16 [1024,64] zero-padded K
__global__ void padcopy_fc1(const float* __restrict__ in, BF* __restrict__ out) {
  int idx = blockIdx.x * 256 + threadIdx.x;   // 1024*64
  int nrow = idx >> 6, k = idx & 63;
  out[idx] = tobf(k < NIN ? in[nrow * NIN + k] : 0.f);
}

// in[K,N] f32 -> out[Npad,Kpad] bf16 (transposed, zero padded) via LDS tile
__global__ __launch_bounds__(256) void transpose_bf(
    const float* __restrict__ in, int K, int N,
    BF* __restrict__ out, int Kpad, int Npad)
{
  __shared__ float tile[32][33];
  int kb = blockIdx.x * 32, nb = blockIdx.y * 32;
  int tx = threadIdx.x & 31, ty = threadIdx.x >> 5;
  for (int i = ty; i < 32; i += 8) {
    int k = kb + tx, n = nb + i;
    tile[i][tx] = (k < K && n < N) ? in[(size_t)k * N + n] : 0.f;
  }
  __syncthreads();
  for (int i = ty; i < 32; i += 8) {
    int n = nb + tx, k = kb + i;
    if (n < Npad && k < Kpad) out[(size_t)n * Kpad + k] = tobf(tile[tx][i]);
  }
}

// softmax(X @ W^T + b) over E=8 experts; thread per (b,e); shfl within lane-groups of 8
__global__ __launch_bounds__(256) void gate8(
    const float* __restrict__ X, int ldx, int K,
    const float* __restrict__ W, const float* __restrict__ bias,
    float* __restrict__ G)
{
  int t = blockIdx.x * 256 + threadIdx.x;
  int b = t >> 3, e = t & 7;
  const float* xp = X + (size_t)b * ldx;
  const float* wp = W + (size_t)e * K;
  float acc = bias[e];
  for (int k = 0; k < K; k++) acc += xp[k] * wp[k];
  float m = acc;
  for (int o = 4; o > 0; o >>= 1) m = fmaxf(m, __shfl_xor(m, o, 32));
  float ex = expf(acc - m);
  float s = ex;
  for (int o = 4; o > 0; o >>= 1) s += __shfl_xor(s, o, 32);
  G[t] = ex / s;
}

// Gumbel softmax over D=64 per (b,c) row; one wave per row (2 elems/lane)
__global__ __launch_bounds__(128) void gumbel_k(
    const float* __restrict__ logits, const float* __restrict__ noise,
    const float* __restrict__ knn,
    float* __restrict__ probs, BF* __restrict__ hard)
{
  int gw = (blockIdx.x * 128 + threadIdx.x) >> 5;   // row = b*C + c
  int lane = threadIdx.x & 31;
  const float s = knn[0];
  size_t base = (size_t)gw * 64;
  float u0 = noise[base + lane];
  float u1 = noise[base + lane + 32];
  float v0 = logits[base + lane]      + (-logf(-logf(s * (u0 - 0.5f) + 0.5f + 1e-20f) + 1e-20f));
  float v1 = logits[base + lane + 32] + (-logf(-logf(s * (u1 - 0.5f) + 0.5f + 1e-20f) + 1e-20f));
  float m = fmaxf(v0, v1);
  for (int o = 16; o > 0; o >>= 1) m = fmaxf(m, __shfl_xor(m, o, 32));
  float e0 = expf(v0 - m), e1 = expf(v1 - m);
  float ss = e0 + e1;
  for (int o = 16; o > 0; o >>= 1) ss += __shfl_xor(ss, o, 32);
  probs[base + lane]      = e0 / ss;
  probs[base + lane + 32] = e1 / ss;
  if (hard) {
    float bv = v0; int bi = lane;
    if (v1 > bv) { bv = v1; bi = lane + 32; }
    for (int o = 16; o > 0; o >>= 1) {
      float ov = __shfl_xor(bv, o, 32);
      int   oi = __shfl_xor(bi, o, 32);
      if (ov > bv || (ov == bv && oi < bi)) { bv = ov; bi = oi; }
    }
    hard[base + lane]      = tobf(lane == bi ? 1.f : 0.f);
    hard[base + lane + 32] = tobf(lane + 32 == bi ? 1.f : 0.f);
  }
}

// GRU cell with h_prev = 0:  h' = (1-z)*n,  gates from gi[B,3H] + bhh
__global__ void gru_elem(const float* __restrict__ gi, const float* __restrict__ bhh,
                         float* __restrict__ Hf, BF* __restrict__ Hbf)
{
  int idx = blockIdx.x * blockDim.x + threadIdx.x;  // BSZ*H
  int b = idx >> 10, h = idx & 1023;
  size_t o = (size_t)b * H3;
  float r = 1.f / (1.f + expf(-(gi[o + h]        + bhh[h])));
  float z = 1.f / (1.f + expf(-(gi[o + 1024 + h] + bhh[1024 + h])));
  float n = tanhf(gi[o + 2048 + h] + r * bhh[2048 + h]);
  float out = (1.f - z) * n;
  Hf[idx] = out;
  if (Hbf) Hbf[idx] = tobf(out);
}

__global__ void fillf(float* __restrict__ p, int n, float v) {
  int i = blockIdx.x * 256 + threadIdx.x;
  if (i < n) p[i] = v;
}

__global__ void copy_y(const float* __restrict__ Y, float* __restrict__ out, int n) {
  int i = blockIdx.x * 256 + threadIdx.x;
  if (i < n) { int b = i / NIN, j = i - b * NIN; out[i] = Y[(size_t)b * 64 + j]; }
}

__global__ __launch_bounds__(256) void sqdiff_part(
    const float* __restrict__ a, const float* __restrict__ b,
    int n, int cols, int ld, float* __restrict__ part)
{
  float s = 0.f;
  for (int i = blockIdx.x * 256 + threadIdx.x; i < n; i += 256 * 256) {
    int row = i / cols, c = i - row * cols;
    size_t o = (size_t)row * ld + c;
    float d = a[o] - b[o];
    s += d * d;
  }
  __shared__ float sh[256];
  sh[threadIdx.x] = s; __syncthreads();
  for (int o = 128; o > 0; o >>= 1) { if (threadIdx.x < o) sh[threadIdx.x] += sh[threadIdx.x + o]; __syncthreads(); }
  if (threadIdx.x == 0) part[blockIdx.x] = sh[0];
}

__global__ __launch_bounds__(256) void red_final(const float* __restrict__ part,
                                                 float inv, float* __restrict__ out)
{
  __shared__ float sh[256];
  sh[threadIdx.x] = part[threadIdx.x]; __syncthreads();
  for (int o = 128; o > 0; o >>= 1) { if (threadIdx.x < o) sh[threadIdx.x] += sh[threadIdx.x + o]; __syncthreads(); }
  if (threadIdx.x == 0) out[0] = sh[0] * inv;
}

// ---------------------------------------------------------------------------
extern "C" void kernel_launch(void* const* d_in, const int* in_sizes, int n_in,
                              void* d_out, int out_size, void* d_ws, size_t ws_size,
                              hipStream_t stream)
{
  (void)in_sizes; (void)n_in; (void)out_size; (void)ws_size;
  const float* x       = (const float*)d_in[0];
  const float* t       = (const float*)d_in[1];
  const float* knn     = (const float*)d_in[2];
  const float* noise_t = (const float*)d_in[3];
  const float* noise_e = (const float*)d_in[4];
  const float* const* P = (const float* const*)(d_in + 5);
  // P index map (params dict order):
  // 0 e_fc1_w 1 e_fc1_b 2 e_ln1_g 3 e_ln1_b 4 e_fc2_w 5 e_fc2_b 6 e_ln2_g 7 e_ln2_b
  // 8 e_fc3_w 9 e_fc3_b 10 e_fc4_w 11 e_fc4_b 12 est_gate_w 13 est_gate_b
  // 14 est_w1 15 est_b1 16 est_w2 17 est_b2 18 est_w3 19 est_b3 20 est_w4 21 est_b4
  // 22 gru_wih0 23 gru_whh0 24 gru_bih0 25 gru_bhh0 26 gru_wih1 27 gru_whh1
  // 28 gru_bih1 29 gru_bhh1 30 d_ln1_g 31 d_ln1_b 32 dmoe_gate_w 33 dmoe_gate_b
  // 34 dmoe_w1 35 dmoe_b1 36 dmoe_w2 37 dmoe_b2 38 dmoe_w3 39 dmoe_b3 40 dmoe_w4 41 dmoe_b4

  uint8_t* wsp = (uint8_t*)d_ws;
  auto af = [&](size_t n) -> float* { float* p = (float*)wsp; wsp += (n * 4 + 255) & ~(size_t)255; return p; };
  auto ab = [&](size_t n) -> BF*    { BF*    p = (BF*)wsp;    wsp += (n * 2 + 255) & ~(size_t)255; return p; };

  float* XV  = af((size_t)BSZ * 64);
  float* TV  = af((size_t)BSZ * 64);
  float* ACT0= af((size_t)BSZ * H3);
  float* TL  = af((size_t)BSZ * CD);
  float* EL  = af((size_t)BSZ * CD);
  float* TP  = af((size_t)BSZ * CD);
  float* EP  = af((size_t)BSZ * CD);
  float* H1  = af((size_t)BSZ * H);
  float* H2  = af((size_t)BSZ * H);
  float* X1  = af((size_t)BSZ * H);
  float* Y   = af((size_t)BSZ * 64);
  float* EG  = af((size_t)BSZ * 8);
  float* DG  = af((size_t)BSZ * 8);
  float* RED = af(256);
  BF* XVB = ab((size_t)BSZ * 64);
  BF* TVB = ab((size_t)BSZ * 64);
  BF* BFA = ab((size_t)BSZ * H);
  BF* BFB = ab((size_t)BSZ * H);
  BF* THB = ab((size_t)BSZ * CD);
  BF* H1B = ab((size_t)BSZ * H);
  BF* X1B = ab((size_t)BSZ * H);
  BF* W1  = ab((size_t)H * KP);
  BF* W2  = ab((size_t)H * H);
  BF* W3  = ab((size_t)H * H);
  BF* W4e = ab((size_t)CD * H);
  BF* WG0 = ab((size_t)H3 * H);
  BF* WG1 = ab((size_t)H3 * H);
  BF* WT  = ab((size_t)H * H);

  auto gemm = [&](const BF* A, int lda, const BF* Bm, int ldb, const float* bias,
                  float* Df, BF* Dbf, int ldd, const float* scale,
                  int N, int K, int nvalid, int act) {
    dim3 g(N / 64, BSZ / 128);
    gemm_bf16<<<g, 128, 0, stream>>>(A, lda, Bm, ldb, bias, Df, Dbf, ldd, scale, K, nvalid, act);
  };
  auto ln = [&](const float* Xp, const float* g_, const float* b_, float* Yf, BF* Ybf, int pre) {
    ln1024<<<BSZ, 256, 0, stream>>>(Xp, 1024, g_, b_, Yf, Ybf, 1024, pre);
  };
  auto trans = [&](const float* in, int K, int N, BF* out, int Kpad, int Npad) {
    dim3 g((Kpad + 31) / 32, (Npad + 31) / 32);
    transpose_bf<<<g, 256, 0, stream>>>(in, K, N, out, Kpad, Npad);
  };

  // ---- prep + one-time weight conversions ------------------------------------
  prep_vec<<<BSZ * 64 / 256, 256, 0, stream>>>(x, t, XV, TV, XVB, TVB);
  padcopy_fc1<<<(H * KP) / 256, 256, 0, stream>>>(P[0], W1);
  conv_bf<<<(H * H) / 256, 256, 0, stream>>>(P[4], W2, H * H);
  conv_bf<<<(H * H) / 256, 256, 0, stream>>>(P[8], W3, H * H);
  conv_bf<<<(CD * H) / 256, 256, 0, stream>>>(P[10], W4e, CD * H);
  conv_bf<<<(H3 * H) / 256, 256, 0, stream>>>(P[22], WG0, H3 * H);
  conv_bf<<<(H3 * H) / 256, 256, 0, stream>>>(P[26], WG1, H3 * H);

  // ---- encoder(t_vec) -> target_logits ---------------------------------------
  gemm(TVB, 64, W1, 64, P[1], ACT0, nullptr, 1024, nullptr, 1024, 64, 1024, ACT_ELU);
  ln(ACT0, P[2], P[3], nullptr, BFA, 0);
  gemm(BFA, 1024, W2, 1024, P[5], ACT0, nullptr, 1024, nullptr, 1024, 1024, 1024, ACT_ELU);
  ln(ACT0, P[6], P[7], nullptr, BFB, 0);
  gemm(BFB, 1024, W3, 1024, P[9], nullptr, BFA, 1024, nullptr, 1024, 1024, 1024, ACT_ELU);
  gemm(BFA, 1024, W4e, 1024, P[11], TL, nullptr, 1024, nullptr, 1024, 1024, 1024, ACT_NONE);

  // target sample: probs + hard one-hot (bf16, feeds GRU)
  gumbel_k<<<(BSZ * NC) / 4, 128, 0, stream>>>(TL, noise_t, knn, TP, THB);

  // ---- estimator MoE(x_vec) -> estimate_logits -------------------------------
  gate8<<<(BSZ * 8) / 256, 256, 0, stream>>>(XV, 64, NIN, P[12], P[13], EG);
  fillf<<<(BSZ * CD) / 256, 256, 0, stream>>>(EL, BSZ * CD, 0.f);
  for (int e = 0; e < 8; e++) {
    trans(P[14] + (size_t)e * NIN * H, NIN, H, WT, KP, H);
    gemm(XVB, 64, WT, KP, P[15] + (size_t)e * H, ACT0, nullptr, 1024, nullptr, 1024, 64, 1024, ACT_MISH);
    ln(ACT0, nullptr, nullptr, nullptr, BFA, 0);
    trans(P[16] + (size_t)e * H * H, H, H, WT, H, H);
    gemm(BFA, 1024, WT, 1024, P[17] + (size_t)e * H, ACT0, nullptr, 1024, nullptr, 1024, 1024, 1024, ACT_MISH);
    ln(ACT0, nullptr, nullptr, nullptr, BFB, 0);
    trans(P[18] + (size_t)e * H * H, H, H, WT, H, H);
    gemm(BFB, 1024, WT, 1024, P[19] + (size_t)e * H, nullptr, BFA, 1024, nullptr, 1024, 1024, 1024, ACT_MISH);
    trans(P[20] + (size_t)e * H * CD, H, CD, WT, H, CD);
    gemm(BFA, 1024, WT, 1024, P[21] + (size_t)e * CD, EL, nullptr, 1024, EG + e, 1024, 1024, 1024, ACT_NONE);
  }
  gumbel_k<<<(BSZ * NC) / 4, 128, 0, stream>>>(EL, noise_e, knn, EP, nullptr);

  // ---- decoder GRU (h0 = 0 -> gh collapses to bhh) ---------------------------
  gemm(THB, 1024, WG0, 1024, P[24], ACT0, nullptr, H3, nullptr, H3, 1024, H3, ACT_NONE);
  gru_elem<<<(BSZ * H) / 256, 256, 0, stream>>>(ACT0, P[25], H1, H1B);
  gemm(H1B, 1024, WG1, 1024, P[28], ACT0, nullptr, H3, nullptr, H3, 1024, H3, ACT_NONE);
  gru_elem<<<(BSZ * H) / 256, 256, 0, stream>>>(ACT0, P[29], H2, nullptr);
  // x1 = LN_affine(elu(h2))
  ln1024<<<BSZ, 256, 0, stream>>>(H2, 1024, P[30], P[31], X1, X1B, 1024, 1);

  // ---- decoder MoE(x1) -> y ---------------------------------------------------
  gate8<<<(BSZ * 8) / 256, 256, 0, stream>>>(X1, 1024, H, P[32], P[33], DG);
  fillf<<<(BSZ * 64) / 256, 256, 0, stream>>>(Y, BSZ * 64, 0.f);
  for (int e = 0; e < 8; e++) {
    trans(P[34] + (size_t)e * H * H, H, H, WT, H, H);
    gemm(X1B, 1024, WT, 1024, P[35] + (size_t)e * H, ACT0, nullptr, 1024, nullptr, 1024, 1024, 1024, ACT_MISH);
    ln(ACT0, nullptr, nullptr, nullptr, BFA, 0);
    trans(P[36] + (size_t)e * H * H, H, H, WT, H, H);
    gemm(BFA, 1024, WT, 1024, P[37] + (size_t)e * H, ACT0, nullptr, 1024, nullptr, 1024, 1024, 1024, ACT_MISH);
    ln(ACT0, nullptr, nullptr, nullptr, BFB, 0);
    trans(P[38] + (size_t)e * H * H, H, H, WT, H, H);
    gemm(BFB, 1024, WT, 1024, P[39] + (size_t)e * H, nullptr, BFA, 1024, nullptr, 1024, 1024, 1024, ACT_MISH);
    trans(P[40] + (size_t)e * H * NIN, H, NIN, WT, H, 64);
    gemm(BFA, 1024, WT, 1024, P[41] + (size_t)e * NIN, Y, nullptr, 64, DG + e, 64, 1024, NIN, ACT_NONE);
  }

  // ---- outputs: y, loss_probs, loss_recon, hidden -----------------------------
  float* out = (float*)d_out;
  const int YN = BSZ * NIN;                         // 143360
  copy_y<<<(YN + 255) / 256, 256, 0, stream>>>(Y, out, YN);
  sqdiff_part<<<256, 256, 0, stream>>>(TP, EP, BSZ * CD, CD, CD, RED);
  red_final<<<1, 256, 0, stream>>>(RED, 1.f / ((float)BSZ * CD), out + YN);
  sqdiff_part<<<256, 256, 0, stream>>>(TV, Y, BSZ * NIN, NIN, 64, RED);
  red_final<<<1, 256, 0, stream>>>(RED, 1.f / ((float)BSZ * NIN), out + YN + 1);
  hipMemcpyAsync(out + YN + 2,            H1, (size_t)BSZ * H * sizeof(float), hipMemcpyDeviceToDevice, stream);
  hipMemcpyAsync(out + YN + 2 + BSZ * H,  H2, (size_t)BSZ * H * sizeof(float), hipMemcpyDeviceToDevice, stream);
}
